// VectorQuantizer_6141803233291
// MI455X (gfx1250) — matrix operand order repaired and checked by reading, not compiled, not run on hardware
//
#include <hip/hip_runtime.h>
#include <hip/hip_bf16.h>

typedef __attribute__((ext_vector_type(16))) __bf16          v16bf;
typedef __attribute__((ext_vector_type(8)))  float           v8f;
typedef __attribute__((ext_vector_type(8)))  unsigned short  us8;
typedef __attribute__((ext_vector_type(16))) unsigned short  us16;

#define CDIM   256
#define KCODES 8192
#define NPTS   16384      /* 16*32*32 */
#define NELEM  4194304    /* 16*256*32*32 */
#define GBLK   16384      /* NELEM/256 */
#define KSPLIT 4
#define KSEG   (KCODES / KSPLIT)   /* 2048 codes per segment   */
#define SEGT   (KSEG / 16)         /* 128 16-row tiles/segment */

__device__ __forceinline__ unsigned short f32_to_bf16(float f) {
  unsigned int u = __float_as_uint(f);
  u += 0x7FFFu + ((u >> 16) & 1u);          // round-to-nearest-even
  return (unsigned short)(u >> 16);
}

__device__ __forceinline__ v16bf makeFrag(us8 lo, us8 hi) {
  us16 u;
#pragma unroll
  for (int i = 0; i < 8; ++i) { u[i] = lo[i]; u[i + 8] = hi[i]; }
  return __builtin_bit_cast(v16bf, u);
}

__device__ __forceinline__ void waitAsync0() {
#if __has_builtin(__builtin_amdgcn_s_wait_asynccnt)
  __builtin_amdgcn_s_wait_asynccnt(0);
#else
  asm volatile("s_wait_asynccnt 0x0" ::: "memory");
#endif
}

// Async DMA: 32 bytes/lane of a codebook tile straight into LDS (no VGPR staging).
// INST_OFFSET is added to BOTH the LDS and global addresses (ISA 08 §4.4), so the
// same address pair serves both 16B halves.
__device__ __forceinline__ void asyncStageTile(unsigned ldsByteAddr,
                                               unsigned long long gAddr) {
  asm volatile(
      "global_load_async_to_lds_b128 %0, %1, off\n\t"
      "global_load_async_to_lds_b128 %0, %1, off offset:4096"
      :: "v"(ldsByteAddr), "v"(gAddr) : "memory");
}

// ---- Kernel 1: z [B,C,HW] fp32 -> zf [B*HW, C] bf16 (LDS-tiled transpose) ----
__global__ void zToBf16T(const float* __restrict__ z, unsigned short* __restrict__ zf) {
  __shared__ float tile[32][33];
  const int b = blockIdx.z;
  const int c0 = blockIdx.y * 32, hw0 = blockIdx.x * 32;
  for (int i = threadIdx.y; i < 32; i += 8)
    tile[i][threadIdx.x] = z[(b * 256 + c0 + i) * 1024 + hw0 + threadIdx.x];
  __syncthreads();
  for (int i = threadIdx.y; i < 32; i += 8) {
    float v = tile[threadIdx.x][i];
    zf[(b * 1024 + hw0 + i) * CDIM + c0 + threadIdx.x] = f32_to_bf16(v);
  }
}

// ---- Kernel 2: codebook fp32 -> bf16, plus fp32 row norms ----
__global__ void embPrep(const float* __restrict__ emb, unsigned short* __restrict__ ebf,
                        float* __restrict__ enorm) {
  __shared__ float red[256];
  const int k = blockIdx.x, t = threadIdx.x;
  float v = emb[k * CDIM + t];
  ebf[k * CDIM + t] = f32_to_bf16(v);
  red[t] = v * v;
  __syncthreads();
  for (int s = 128; s > 0; s >>= 1) { if (t < s) red[t] += red[t + s]; __syncthreads(); }
  if (t == 0) enorm[k] = red[0];
}

// ---- Kernel 3: WMMA distance + per-segment argmin ----
// blockIdx = rowgroup (256 z rows) x k-segment (2048 codes).
// 8 waves/block; each wave owns 32 z rows as register-resident B fragments and
// issues 2 WMMAs per codebook A-fragment. Codebook tiles (8KB) are double-
// buffered in LDS via async global->LDS DMA (ASYNCcnt), one barrier per tile.
__global__ __launch_bounds__(256) void vqArgmin(const unsigned short* __restrict__ zf,
                                                const unsigned short* __restrict__ ebf,
                                                const float* __restrict__ enorm,
                                                float* __restrict__ segV,
                                                int* __restrict__ segI) {
  __shared__ __align__(16) unsigned short tile[2][16 * CDIM];   // 16 KB
  __shared__ float eN[KSEG];                                    //  8 KB
  const int t = threadIdx.x;
  const int wave = t >> 5, lane = t & 31;
  const int lrow = lane & 15, half = lane >> 4;
  const int grp = blockIdx.x / KSPLIT;        // row group (256 rows)
  const int seg = blockIdx.x % KSPLIT;        // code segment
  const int kbase = seg * KSEG;
  const unsigned short* ebfS = ebf + (size_t)kbase * CDIM;

  for (int i = t; i < KSEG; i += 256) eN[i] = enorm[kbase + i];

  // two z slabs -> register-resident B fragments (B[c,n] = z_n[c]; lane = column n)
  const int n0 = grp * 256 + wave * 32;
  v16bf bFrag[2][8];
#pragma unroll
  for (int sl = 0; sl < 2; ++sl) {
    const int zrow = n0 + sl * 16 + lrow;
#pragma unroll
    for (int s = 0; s < 8; ++s) {
      const us8* p = (const us8*)(zf + zrow * CDIM + s * 32 + half * 16);
      bFrag[sl][s] = makeFrag(p[0], p[1]);
    }
  }

  const unsigned ldsBase = (unsigned)(size_t)&tile[0][0] + (unsigned)(t * 16);
  const unsigned long long gBase =
      (unsigned long long)(size_t)ebfS + (unsigned long long)(t * 16);

  // async-stage codebook tile 0 of this segment
  asyncStageTile(ldsBase, gBase);
  waitAsync0();
  __syncthreads();

  float minv0 = 3.402823466e38f, minv1 = 3.402823466e38f;
  int   mini0 = 0, mini1 = 0;

  for (int kt = 0; kt < SEGT; ++kt) {
    const int buf = kt & 1;
    if (kt + 1 < SEGT)          // DMA next tile into the other buffer
      asyncStageTile(ldsBase + (unsigned)((buf ^ 1) * 8192),
                     gBase + (unsigned long long)(kt + 1) * (16 * CDIM * 2));
    if (kt + 2 < SEGT)          // gfx1250 global_prefetch for tile after next
      __builtin_prefetch(ebfS + (kt + 2) * 16 * CDIM + t * 16, 0, 3);

    v8f a0 = {}, a1 = {};
#pragma unroll
    for (int s = 0; s < 8; ++s) {
      // A fragment (codebook): lane holds emb row lrow of the tile,
      // K chunks {0..15} (half=0) / {16..31} (half=1)
      const us8* pl = (const us8*)&tile[buf][lrow * CDIM + s * 32 + half * 16];
      const us8* ph = (const us8*)&tile[buf][lrow * CDIM + s * 32 + 8 + half * 16];
      v16bf aF = makeFrag(*pl, *ph);
      a0 = __builtin_amdgcn_wmma_f32_16x16x32_bf16(false, aF, false, bFrag[0][s],
                                                   (short)0, a0, false, false);
      a1 = __builtin_amdgcn_wmma_f32_16x16x32_bf16(false, aF, false, bFrag[1][s],
                                                   (short)0, a1, false, false);
    }

    // D layout: lane = z column (lrow); VGPR j = emb row half*8+j of this tile
    const int mloc = kt * 16 + half * 8;
#pragma unroll
    for (int j = 0; j < 8; ++j) {
      const float en = eN[mloc + j];
      const float s0 = en - 2.0f * a0[j];
      const float s1 = en - 2.0f * a1[j];
      const int gidx = kbase + mloc + j;
      if (s0 < minv0) { minv0 = s0; mini0 = gidx; }
      if (s1 < minv1) { minv1 = s1; mini1 = gidx; }
    }

    waitAsync0();               // this wave's DMA writes have landed in LDS
    __syncthreads();            // all waves' tiles visible; prev reads retired
  }

  // combine lane L (rows 0..7) with lane L+16 (rows 8..15); first-index tie-break
  {
    float ov = __shfl_xor(minv0, 16, 32);
    int   oi = __shfl_xor(mini0, 16, 32);
    if (ov < minv0 || (ov == minv0 && oi < mini0)) { minv0 = ov; mini0 = oi; }
    ov = __shfl_xor(minv1, 16, 32);
    oi = __shfl_xor(mini1, 16, 32);
    if (ov < minv1 || (ov == minv1 && oi < mini1)) { minv1 = ov; mini1 = oi; }
  }
  if (half == 0) {
    segV[seg * NPTS + n0 + lrow]      = minv0;
    segI[seg * NPTS + n0 + lrow]      = mini0;
    segV[seg * NPTS + n0 + 16 + lrow] = minv1;
    segI[seg * NPTS + n0 + 16 + lrow] = mini1;
  }
}

// ---- Kernel 4: combine per-segment argmins (segments ascending => tie-break ok) ----
__global__ void vqCombine(const float* __restrict__ segV, const int* __restrict__ segI,
                          int* __restrict__ minIdx, float* __restrict__ outIdxF) {
  const int n = blockIdx.x * 256 + threadIdx.x;
  float bv = segV[n];
  int   bi = segI[n];
  for (int s = 1; s < KSPLIT; ++s) {
    const float v = segV[s * NPTS + n];
    const int   i = segI[s * NPTS + n];
    if (v < bv || (v == bv && i < bi)) { bv = v; bi = i; }
  }
  minIdx[n] = bi;
  outIdxF[n] = (float)bi;
}

// ---- Kernel 5: gather z_q (exact fp32) + per-block loss partials ----
__global__ void vqGather(const float* __restrict__ z, const float* __restrict__ emb,
                         const int* __restrict__ minIdx, float* __restrict__ zq,
                         float* __restrict__ partial) {
  __shared__ float red[256];
  const int t = threadIdx.x;
  const int gid = blockIdx.x * 256 + t;         // [b][c][hw] order
  const int hw = gid & 1023;
  const int c  = (gid >> 10) & 255;
  const int b  = gid >> 18;
  const int n  = b * 1024 + hw;
  const int idx = minIdx[n];
  const float e  = emb[idx * CDIM + c];
  const float zv = z[gid];
  zq[gid] = e;                                   // straight-through fwd value
  const float d = e - zv;
  red[t] = d * d;
  __syncthreads();
  for (int s = 128; s > 0; s >>= 1) { if (t < s) red[t] += red[t + s]; __syncthreads(); }
  if (t == 0) partial[blockIdx.x] = red[0];
}

// ---- Kernel 6: deterministic loss reduction; loss = (1+BETA)*mean(diff^2) ----
__global__ void vqLoss(const float* __restrict__ partial, float* __restrict__ outLoss) {
  __shared__ float red[256];
  const int t = threadIdx.x;
  float s = 0.0f;
  for (int i = t; i < GBLK; i += 256) s += partial[i];
  red[t] = s;
  __syncthreads();
  for (int k = 128; k > 0; k >>= 1) { if (t < k) red[t] += red[t + k]; __syncthreads(); }
  if (t == 0) *outLoss = 1.25f * red[0] / (float)NELEM;
}

extern "C" void kernel_launch(void* const* d_in, const int* in_sizes, int n_in,
                              void* d_out, int out_size, void* d_ws, size_t ws_size,
                              hipStream_t stream) {
  const float* z   = (const float*)d_in[0];   // [16,256,32,32]
  const float* emb = (const float*)d_in[1];   // [8192,256]
  float* out = (float*)d_out;

  char* ws = (char*)d_ws;
  unsigned short* zf      = (unsigned short*)(ws);              //  8 MB
  unsigned short* ebf     = (unsigned short*)(ws + 8388608);    //  4 MB
  float*          enorm   = (float*)(ws + 12582912);            // 32 KB
  int*            minIdx  = (int*)(ws + 12615680);              // 64 KB
  float*          partial = (float*)(ws + 12681216);            // 64 KB
  float*          segV    = (float*)(ws + 12746752);            // 256 KB
  int*            segI    = (int*)(ws + 13008896);              // 256 KB

  float* zq      = out;                 // [16,256,32,32]
  float* outLoss = out + NELEM;         // scalar
  float* outIdx  = out + NELEM + 1;     // [16384] indices as float

  zToBf16T <<<dim3(32, 8, 16), dim3(32, 8), 0, stream>>>(z, zf);
  embPrep  <<<KCODES, 256, 0, stream>>>(emb, ebf, enorm);
  vqArgmin <<<(NPTS / 256) * KSPLIT, 256, 0, stream>>>(zf, ebf, enorm, segV, segI);
  vqCombine<<<NPTS / 256, 256, 0, stream>>>(segV, segI, minIdx, outIdx);
  vqGather <<<GBLK, 256, 0, stream>>>(z, emb, minIdx, zq, partial);
  vqLoss   <<<1, 256, 0, stream>>>(partial, outLoss);
}
